// RewardForcingSelfAttention_45200235823673
// MI455X (gfx1250) — matrix operand order repaired and checked by reading, not compiled, not run on hardware
//
#include <hip/hip_runtime.h>
#include <cstdint>
#include <cstddef>

// ---------------- problem constants (from the reference) ----------------
#define SEQ_Q   1560          // S
#define SEQ_KV  9360          // TOTAL_SINK + ACTUAL_KV
#define NH      12
#define HD      128
#define DIMF    1536
#define KV_CHUNKS 293         // ceil(9360/32)
#define Q_FRAME 5             // Q_START_FRAME

// -------- CDNA5 async global->LDS path (guarded; falls back if absent) --------
#ifndef __has_builtin
#define __has_builtin(x) 0
#endif
#if defined(__gfx1250__) && __has_builtin(__builtin_amdgcn_global_load_async_to_lds_b128)
#define ASYNC_LDS 1
#endif

#ifdef ASYNC_LDS
typedef int vi4 __attribute__((ext_vector_type(4)));
typedef __attribute__((address_space(1))) vi4* as1_vi4p;   // global int4*
typedef __attribute__((address_space(3))) vi4* as3_vi4p;   // LDS int4*

__device__ __forceinline__ void async_copy_b128(const void* g, void* l) {
  // integer round-trips: AS1 value == flat address; low 32 bits of a generic
  // LDS pointer are the DS byte offset.
  __builtin_amdgcn_global_load_async_to_lds_b128(
      (as1_vi4p)(unsigned long long)g,
      (as3_vi4p)(unsigned)(unsigned long long)l,
      0, 0);
}
__device__ __forceinline__ void async_wait0() {
#if __has_builtin(__builtin_amdgcn_s_wait_asynccnt)
  __builtin_amdgcn_s_wait_asynccnt(0);
#else
  asm volatile("s_wait_asynccnt 0x0" ::: "memory");
#endif
}
#endif

typedef __attribute__((ext_vector_type(16))) __bf16 v16bf;
typedef __attribute__((ext_vector_type(8)))  float  v8f;

struct __attribute__((aligned(16))) U4 { unsigned int x, y, z, w; };

union Frag16 { v16bf v; U4 u[2]; };
union FragC  { v8f  v; float f[8]; };

__device__ __forceinline__ unsigned short f2bf(float x) {
  union { float f; unsigned int u; } c; c.f = x;
  unsigned int r = c.u + 0x7FFFu + ((c.u >> 16) & 1u);   // round-to-nearest-even
  return (unsigned short)(r >> 16);
}

// A-operand fragment (16x32 MxK, 16-bit): lane m = lane&15, hi = lane>>4
//   VGPR0..3 hold K = hi*8 .. hi*8+7 ; VGPR4..7 hold K = 16+hi*8 .. 23+hi*8
__device__ __forceinline__ void frag_a(Frag16& f, const unsigned short* row, int hi) {
  f.u[0] = *(const U4*)(row + hi * 8);
  f.u[1] = *(const U4*)(row + 16 + hi * 8);
}
// B-operand fragment (32x16 KxN, 16-bit) from (N,K)-major storage:
//   lane n = lane&15, hi = lane>>4 ; VGPR0..3 hold K = hi*8 .. hi*8+7 ; VGPR4..7 hold K = 16+hi*8 .. 23+hi*8
__device__ __forceinline__ void frag_b(Frag16& f, const unsigned short* row, int hi) {
  f.u[0] = *(const U4*)(row + hi * 8);
  f.u[1] = *(const U4*)(row + 16 + hi * 8);
}
__device__ __forceinline__ void frag_zero(Frag16& f) {
  f.u[0] = U4{0u,0u,0u,0u}; f.u[1] = U4{0u,0u,0u,0u};
}

// ---------------- generic GEMM: C = A(MxK) @ W(KxN) + bias ----------------
// fp32 in, bf16 WMMA, fp32 out. Block = 4 waves, tile 64x64, K-step 32.
__global__ __launch_bounds__(128)
void gemm_bias_kernel(const float* __restrict__ A, const float* __restrict__ W,
                      const float* __restrict__ bias, float* __restrict__ C,
                      int M, int N, int K) {
  __shared__ unsigned short As[64 * 40];   // [m][k], padded stride 40
  __shared__ unsigned short Bs[64 * 40];   // [n][k] (transposed), padded
  const int tid = threadIdx.x, lane = tid & 31, wave = tid >> 5;
  const int nl = lane & 15, hi = lane >> 4;
  const int m0 = blockIdx.x * 64, n0 = blockIdx.y * 64;

  FragC acc[4];
  for (int t = 0; t < 4; ++t)
    for (int r = 0; r < 8; ++r) acc[t].f[r] = 0.f;

  for (int k0 = 0; k0 < K; k0 += 32) {
    for (int i = 0; i < 16; ++i) {
      int idx = i * 128 + tid, r = idx >> 5, c = idx & 31;
      int gr = m0 + r;
      float v = (gr < M) ? A[(size_t)gr * K + k0 + c] : 0.f;
      As[r * 40 + c] = f2bf(v);
    }
    for (int i = 0; i < 16; ++i) {
      int idx = i * 128 + tid, r = idx >> 6, c = idx & 63;
      float v = W[(size_t)(k0 + r) * N + n0 + c];
      Bs[c * 40 + r] = f2bf(v);
    }
    __syncthreads();
    if (k0 + 32 < K) {                     // prefetch next K-tile (global_prefetch)
      int pr = m0 + (tid >> 1); if (pr >= M) pr = M - 1;
      __builtin_prefetch(&A[(size_t)pr * K + k0 + 32], 0, 1);
      __builtin_prefetch(&W[(size_t)(k0 + 32 + (tid & 31)) * N + n0], 0, 1);
    }
    Frag16 fa; frag_a(fa, &As[(wave * 16 + nl) * 40], hi);
    for (int t = 0; t < 4; ++t) {
      Frag16 fb; frag_b(fb, &Bs[(t * 16 + nl) * 40], hi);
      acc[t].v = __builtin_amdgcn_wmma_f32_16x16x32_bf16(false, fa.v, false, fb.v,
                                                         (short)0, acc[t].v, false, false);
    }
    __syncthreads();
  }
  for (int t = 0; t < 4; ++t) {
    int col = n0 + t * 16 + nl;
    float bb = bias[col];
    for (int r = 0; r < 8; ++r) {
      int row = m0 + wave * 16 + r + 8 * hi;
      if (row < M) C[(size_t)row * N + col] = acc[t].f[r] + bb;
    }
  }
}

// ---------------- row rms (rsqrt of mean square) for q_lin / k_lin ----------------
__global__ __launch_bounds__(256)
void rownorm_kernel(const float* __restrict__ qlin, const float* __restrict__ klin,
                    float* __restrict__ rinv) {
  int row = blockIdx.x, which = blockIdx.y;
  const float* src = (which == 0 ? qlin : klin) + (size_t)row * DIMF;
  float s = 0.f;
  for (int i = threadIdx.x; i < DIMF; i += 256) { float v = src[i]; s += v * v; }
  s += __shfl_xor(s, 1, 32);  s += __shfl_xor(s, 2, 32);
  s += __shfl_xor(s, 4, 32);  s += __shfl_xor(s, 8, 32);  s += __shfl_xor(s, 16, 32);
  __shared__ float red[8];
  if ((threadIdx.x & 31) == 0) red[threadIdx.x >> 5] = s;
  __syncthreads();
  if (threadIdx.x == 0) {
    float v = 0.f;
    for (int i = 0; i < 8; ++i) v += red[i];
    rinv[which * SEQ_Q + row] = rsqrtf(v / (float)DIMF + 1e-6f);
  }
}

// ---------------- build bf16 Q (rmsnorm * gq + RoPE), layout [head][s][128] -------
__global__ __launch_bounds__(256)
void build_q_kernel(const float* __restrict__ qlin, const float* __restrict__ rinv,
                    const float* __restrict__ gq,
                    const float* __restrict__ fcos, const float* __restrict__ fsin,
                    unsigned short* __restrict__ Qb) {
  int idx = blockIdx.x * 256 + threadIdx.x;        // over SEQ_Q * NH * 64 pairs
  if (idx >= SEQ_Q * NH * 64) return;
  int c = idx & 63;
  int n = (idx >> 6) % NH;
  int s = idx / (64 * NH);
  int tr = (c < 22) ? Q_FRAME : ((c < 43) ? (s / 52) : (s % 52));
  float cs = fcos[tr * 64 + c], sn = fsin[tr * 64 + c];
  size_t ib = (size_t)s * DIMF + n * HD + 2 * c;
  float rz = rinv[s];
  float a = qlin[ib]     * rz * gq[n * HD + 2 * c];
  float b = qlin[ib + 1] * rz * gq[n * HD + 2 * c + 1];
  size_t ob = ((size_t)n * SEQ_Q + s) * HD + 2 * c;
  Qb[ob]     = f2bf(a * cs - b * sn);
  Qb[ob + 1] = f2bf(a * sn + b * cs);
}

// ------- build bf16 K_comb (RoPE'd) and V_comb from old cache + new k/v ----------
// k_comb[m] mapping (from the roll/evict/sink algebra):
//   m <  1560 : 0.999*cache[m] + 0.001*cache[9360+m]
//   m <  7800 : cache[m + 3120]
//   else      : new row (m - 7800)
__global__ __launch_bounds__(256)
void build_kv_kernel(const float* __restrict__ klin, const float* __restrict__ vlin,
                     const float* __restrict__ rinvk, const float* __restrict__ gk,
                     const float* __restrict__ cache_k, const float* __restrict__ cache_v,
                     const float* __restrict__ fcos, const float* __restrict__ fsin,
                     unsigned short* __restrict__ Kb, unsigned short* __restrict__ Vb) {
  int idx = blockIdx.x * 256 + threadIdx.x;        // over SEQ_KV * NH * 64 pairs
  if (idx >= SEQ_KV * NH * 64) return;
  int c = idx & 63;
  int n = (idx >> 6) % NH;
  int m = idx / (64 * NH);
  float ka, kb2, va, vb2;
  if (m < 1560) {
    size_t i0 = ((size_t)m * NH + n) * HD + 2 * c;
    size_t i1 = ((size_t)(9360 + m) * NH + n) * HD + 2 * c;
    ka  = 0.999f * cache_k[i0]     + 0.001f * cache_k[i1];
    kb2 = 0.999f * cache_k[i0 + 1] + 0.001f * cache_k[i1 + 1];
    va  = 0.999f * cache_v[i0]     + 0.001f * cache_v[i1];
    vb2 = 0.999f * cache_v[i0 + 1] + 0.001f * cache_v[i1 + 1];
  } else if (m < 7800) {
    size_t i0 = ((size_t)(m + 3120) * NH + n) * HD + 2 * c;
    ka = cache_k[i0]; kb2 = cache_k[i0 + 1];
    va = cache_v[i0]; vb2 = cache_v[i0 + 1];
  } else {
    int row = m - 7800;
    size_t i0 = (size_t)row * DIMF + n * HD + 2 * c;
    float rz = rinvk[row];
    ka  = klin[i0]     * rz * gk[n * HD + 2 * c];
    kb2 = klin[i0 + 1] * rz * gk[n * HD + 2 * c + 1];
    va = vlin[i0]; vb2 = vlin[i0 + 1];
  }
  int f = m / 1560, p = m % 1560;
  int tr = (c < 22) ? f : ((c < 43) ? (p / 52) : (p % 52));
  float cs = fcos[tr * 64 + c], sn = fsin[tr * 64 + c];
  size_t ob = ((size_t)n * SEQ_KV + m) * HD + 2 * c;
  Kb[ob]     = f2bf(ka * cs - kb2 * sn);
  Kb[ob + 1] = f2bf(ka * sn + kb2 * cs);
  Vb[ob]     = f2bf(va);
  Vb[ob + 1] = f2bf(vb2);
}

// ---------------- flash attention, one head per blockIdx.y ----------------
// 4 waves; each wave owns 16 Q rows; KV streamed in 32-row chunks.
// K chunk staged ONCE per block into LDS via async global->LDS (natural
// [kv][chan] layout == (N,K)-major for the B operand); V chunk staged
// transposed via ds stores (async copy cannot transpose 16-bit lanes).
__global__ __launch_bounds__(128)
void flash_attn_kernel(const unsigned short* __restrict__ Qb,
                       const unsigned short* __restrict__ Kb,
                       const unsigned short* __restrict__ Vb,
                       float* __restrict__ O) {
  __shared__ unsigned short Ks[32 * 136];       // [kv][chan], 272B stride (bank-spread)
  __shared__ unsigned short Vt[HD * 40];        // [chan][kv], padded
  __shared__ unsigned short Pt[4 * 16 * 40];    // per-wave P tile [qrow][kv]
  const int head = blockIdx.y;
  const int tid = threadIdx.x, lane = tid & 31, wave = tid >> 5;
  const int nl = lane & 15, hi = lane >> 4;
  const int row0 = blockIdx.x * 64 + wave * 16;
  const unsigned short* Qh = Qb + (size_t)head * SEQ_Q * HD;
  const unsigned short* Kh = Kb + (size_t)head * SEQ_KV * HD;
  const unsigned short* Vh = Vb + (size_t)head * SEQ_KV * HD;

  Frag16 fq[4];
  {
    int qrow = row0 + nl;                       // per-lane guard (loads only)
    if (qrow < SEQ_Q) {
      const unsigned short* qp = Qh + (size_t)qrow * HD;
      for (int c = 0; c < 4; ++c) frag_a(fq[c], qp + c * 32, hi);
    } else {
      for (int c = 0; c < 4; ++c) frag_zero(fq[c]);
    }
  }

  float mrow[8], lrow[8];
  FragC oacc[8];
  for (int r = 0; r < 8; ++r) { mrow[r] = -1e30f; lrow[r] = 0.f; }
  for (int t = 0; t < 8; ++t)
    for (int r = 0; r < 8; ++r) oacc[t].f[r] = 0.f;

  const float scale = 0.08838834764831845f;     // 1/sqrt(128)

  for (int j = 0; j < KV_CHUNKS; ++j) {
    const int kv0 = j * 32;

    // ---- stage K chunk (32x128 bf16) and V chunk (transposed) ----
    __syncthreads();                            // prior iteration's LDS reads done
#ifdef ASYNC_LDS
    for (int i = 0; i < 4; ++i) {               // 512 x 16B segments / 128 threads
      int seg = i * 128 + tid;
      int kv = seg >> 4, sg = (seg & 15) * 8;   // 8 bf16 per segment
      int krow = kv0 + kv;
      if (krow < SEQ_KV)
        async_copy_b128(Kh + (size_t)krow * HD + sg, &Ks[kv * 136 + sg]);
    }
#else
    for (int i = 0; i < 4; ++i) {
      int seg = i * 128 + tid;
      int kv = seg >> 4, sg = (seg & 15) * 8;
      int krow = kv0 + kv;
      if (krow < SEQ_KV)
        *(U4*)&Ks[kv * 136 + sg] = *(const U4*)(Kh + (size_t)krow * HD + sg);
    }
#endif
    for (int i = 0; i < 4; ++i) {               // V: b128 loads, transposed ds stores
      int seg = i * 128 + tid;
      int kv = seg >> 4, ch8 = (seg & 15) * 8;
      int krow = kv0 + kv;
      U4 d = U4{0u,0u,0u,0u};
      if (krow < SEQ_KV) d = *(const U4*)(Vh + (size_t)krow * HD + ch8);
      unsigned int w[4] = {d.x, d.y, d.z, d.w};
      for (int q = 0; q < 4; ++q) {
        Vt[(ch8 + 2 * q)     * 40 + kv] = (unsigned short)(w[q] & 0xFFFFu);
        Vt[(ch8 + 2 * q + 1) * 40 + kv] = (unsigned short)(w[q] >> 16);
      }
    }
#ifdef ASYNC_LDS
    async_wait0();                              // my async copies landed in LDS
#endif
    __syncthreads();                            // everyone's staging visible

    // ---- S = Q K^T from LDS (subtile guard is wave-uniform: multiples of 16) ----
    FragC s[2];
    for (int t = 0; t < 2; ++t) {
      for (int r = 0; r < 8; ++r) s[t].f[r] = 0.f;
      if (kv0 + t * 16 < SEQ_KV) {
        const unsigned short* kp = &Ks[(t * 16 + nl) * 136];
        for (int c = 0; c < 4; ++c) {
          Frag16 fk; frag_b(fk, kp + c * 32, hi);
          s[t].v = __builtin_amdgcn_wmma_f32_16x16x32_bf16(false, fq[c].v, false, fk.v,
                                                           (short)0, s[t].v, false, false);
        }
      }
    }
    // ---- scale + OOB mask ----
    for (int t = 0; t < 2; ++t) {
      bool oob = (kv0 + t * 16 + nl) >= SEQ_KV;
      for (int r = 0; r < 8; ++r)
        s[t].f[r] = oob ? -1e30f : s[t].f[r] * scale;
    }
    // ---- online softmax (rows live across 16 lanes of each half-wave) ----
    float sc[8];
    for (int r = 0; r < 8; ++r) {
      float v = fmaxf(s[0].f[r], s[1].f[r]);
      v = fmaxf(v, __shfl_xor(v, 1, 32));
      v = fmaxf(v, __shfl_xor(v, 2, 32));
      v = fmaxf(v, __shfl_xor(v, 4, 32));
      v = fmaxf(v, __shfl_xor(v, 8, 32));
      float nm = fmaxf(mrow[r], v);
      sc[r] = __expf(mrow[r] - nm);
      mrow[r] = nm;
    }
    for (int t = 0; t < 2; ++t)
      for (int r = 0; r < 8; ++r)
        s[t].f[r] = __expf(s[t].f[r] - mrow[r]);
    for (int r = 0; r < 8; ++r) {
      float v = s[0].f[r] + s[1].f[r];
      v += __shfl_xor(v, 1, 32);
      v += __shfl_xor(v, 2, 32);
      v += __shfl_xor(v, 4, 32);
      v += __shfl_xor(v, 8, 32);
      lrow[r] = lrow[r] * sc[r] + v;
    }
    for (int t = 0; t < 8; ++t)
      for (int r = 0; r < 8; ++r) oacc[t].f[r] *= sc[r];

    // ---- P: C-layout -> per-wave LDS tile -> A-layout (same-wave, in-order DS) ----
    for (int t = 0; t < 2; ++t)
      for (int r = 0; r < 8; ++r)
        Pt[(wave * 16 + r + 8 * hi) * 40 + t * 16 + nl] = f2bf(s[t].f[r]);
    Frag16 fp; frag_a(fp, &Pt[(wave * 16 + nl) * 40], hi);
    // ---- O += P @ V ----
    for (int t = 0; t < 8; ++t) {
      Frag16 fv; frag_b(fv, &Vt[(t * 16 + nl) * 40], hi);
      oacc[t].v = __builtin_amdgcn_wmma_f32_16x16x32_bf16(false, fp.v, false, fv.v,
                                                          (short)0, oacc[t].v, false, false);
    }
  }
  // ---- normalize and write O[s][head*128+ch] (fp32, feeds final GEMM) ----
  for (int t = 0; t < 8; ++t) {
    int ch = t * 16 + nl;
    for (int r = 0; r < 8; ++r) {
      int row = row0 + r + 8 * hi;
      if (row < SEQ_Q)
        O[(size_t)row * (NH * HD) + head * HD + ch] = oacc[t].f[r] / lrow[r];
    }
  }
}

// ---------------- host-side launcher ----------------
extern "C" void kernel_launch(void* const* d_in, const int* in_sizes, int n_in,
                              void* d_out, int out_size, void* d_ws, size_t ws_size,
                              hipStream_t stream) {
  (void)in_sizes; (void)n_in; (void)out_size; (void)ws_size;
  const float* x    = (const float*)d_in[0];
  const float* Wq   = (const float*)d_in[1];
  const float* bq   = (const float*)d_in[2];
  const float* Wk   = (const float*)d_in[3];
  const float* bk   = (const float*)d_in[4];
  const float* Wv   = (const float*)d_in[5];
  const float* bv   = (const float*)d_in[6];
  const float* Wo   = (const float*)d_in[7];
  const float* bo   = (const float*)d_in[8];
  const float* gq   = (const float*)d_in[9];
  const float* gk   = (const float*)d_in[10];
  const float* ck   = (const float*)d_in[11];
  const float* cv   = (const float*)d_in[12];
  const float* fcos = (const float*)d_in[13];
  const float* fsin = (const float*)d_in[14];
  float* out = (float*)d_out;

  char* ws = (char*)d_ws;
  auto alignup = [](size_t v) { return (v + 255) & ~(size_t)255; };
  const size_t LIN = (size_t)SEQ_Q * DIMF;
  size_t off = 0;
  float* q_lin = (float*)(ws + off); off = alignup(off + LIN * 4);
  float* k_lin = (float*)(ws + off); off = alignup(off + LIN * 4);
  float* v_lin = (float*)(ws + off); off = alignup(off + LIN * 4);
  float* o_buf = (float*)(ws + off); off = alignup(off + LIN * 4);
  float* rinv  = (float*)(ws + off); off = alignup(off + 2 * SEQ_Q * 4);
  unsigned short* q_bf = (unsigned short*)(ws + off);
  off = alignup(off + (size_t)NH * SEQ_Q * HD * 2);
  unsigned short* k_bf = (unsigned short*)(ws + off);
  off = alignup(off + (size_t)NH * SEQ_KV * HD * 2);
  unsigned short* v_bf = (unsigned short*)(ws + off);

  dim3 blk(128);
  dim3 gGemm((SEQ_Q + 63) / 64, DIMF / 64);

  gemm_bias_kernel<<<gGemm, blk, 0, stream>>>(x, Wq, bq, q_lin, SEQ_Q, DIMF, DIMF);
  gemm_bias_kernel<<<gGemm, blk, 0, stream>>>(x, Wk, bk, k_lin, SEQ_Q, DIMF, DIMF);
  gemm_bias_kernel<<<gGemm, blk, 0, stream>>>(x, Wv, bv, v_lin, SEQ_Q, DIMF, DIMF);

  rownorm_kernel<<<dim3(SEQ_Q, 2), 256, 0, stream>>>(q_lin, k_lin, rinv);

  build_q_kernel<<<(SEQ_Q * NH * 64 + 255) / 256, 256, 0, stream>>>(
      q_lin, rinv, gq, fcos, fsin, q_bf);
  build_kv_kernel<<<(SEQ_KV * NH * 64 + 255) / 256, 256, 0, stream>>>(
      k_lin, v_lin, rinv + SEQ_Q, gk, ck, cv, fcos, fsin, k_bf, v_bf);

  flash_attn_kernel<<<dim3((SEQ_Q + 63) / 64, NH), blk, 0, stream>>>(
      q_bf, k_bf, v_bf, o_buf);

  gemm_bias_kernel<<<gGemm, blk, 0, stream>>>(o_buf, Wo, bo, out, SEQ_Q, DIMF, DIMF);
}
